// MoE_66915590472084
// MI455X (gfx1250) — matrix operand order repaired and checked by reading, not compiled, not run on hardware
//
#include <hip/hip_runtime.h>
#include <hip/hip_bf16.h>

#define DDIM 1024
#define IDIM 512
#define NEXP 32
#define KSEL 4
#define NTOK 4096
#define CAP  1024   // per-expert capacity

typedef __bf16 bf16;
typedef __bf16 v8bf  __attribute__((ext_vector_type(8)));
typedef __bf16 v16bf __attribute__((ext_vector_type(16)));
typedef float  v8f   __attribute__((ext_vector_type(8)));
typedef unsigned int v4u __attribute__((ext_vector_type(4)));
typedef int v8i __attribute__((ext_vector_type(8)));
typedef int v4i __attribute__((ext_vector_type(4)));

__device__ __forceinline__ v8f wmma_bf16(v16bf a, v16bf b, v8f c) {
  // (neg_a, A, neg_b, B, c_mod, C, reuse_a, reuse_b)
  return __builtin_amdgcn_wmma_f32_16x16x32_bf16(false, a, false, b, (short)0, c,
                                                 false, false);
}

__device__ __forceinline__ v16bf cat8(v8bf lo, v8bf hi) {
  return __builtin_shufflevector(lo, hi, 0, 1, 2, 3, 4, 5, 6, 7,
                                 8, 9, 10, 11, 12, 13, 14, 15);
}

// ---------------------------------------------------------------------------
// TDM: DMA one [128 rows x 32 elems] bf16 tile (row-contiguous in x, stride
// between rows) straight into LDS with hardware padding: after every 64B row
// (pad_interval code 3 = 16 DWORDs) insert 16B (pad_amount code 3 = 4 DWORDs)
// -> LDS row pitch 80B == LDP(40) elems, matching the fragment loader.
// This toolchain exposes the 6-arg builtin:
//   (uint32x4 g0, int32x8 g1, int32x4 g2, int32x4 g3, int32x8, i32 cpol)
// g2/g3 (and the trailing group) are only needed for >2-D tensors -> zeros.
// ---------------------------------------------------------------------------
__device__ __forceinline__ void tdm_load_tile_2d(unsigned lds_off,
                                                 const void* gaddr,
                                                 int stride_elems) {
  const unsigned long long ga = (unsigned long long)gaddr;
  v4u g0;
  g0[0] = 1u;                                  // count=1 (valid user D#)
  g0[1] = lds_off;                             // lds_addr (bytes)
  g0[2] = (unsigned)ga;                        // global_addr[31:0]
  g0[3] = (unsigned)(ga >> 32) | (2u << 30);   // global_addr[56:32] | type=2
  v8i g1;
  g1[0] = (int)((1u << 16) |                   // data_size = 2 bytes
                (1u << 20) |                   // pad_enable
                (3u << 22) |                   // pad_interval: 16 DWORDs
                (3u << 25));                   // pad_amount: 4 DWORDs
  g1[1] = 0;                                   // bar_addr=0 | tensor_dim0 lo16
  g1[2] = 0x4000;                              // tensor_dim0 hi (1<<30) | dim1 lo
  g1[3] = (int)(0x4000u | (32u << 16));        // tensor_dim1 hi | tile_dim0=32
  g1[4] = 128;                                 // tile_dim1=128, tile_dim2=0
  g1[5] = stride_elems;                        // tensor_dim0_stride (elems)
  g1[6] = 0;
  g1[7] = 0;
  const v4i gz4 = {0, 0, 0, 0};
  const v8i gz8 = {0, 0, 0, 0, 0, 0, 0, 0};
  __builtin_amdgcn_tensor_load_to_lds(g0, g1, gz4, gz4, gz8, 0);
}

// ---------------------------------------------------------------------------
// Router: scores = sigmoid(X @ gate_w^T); top-4 on (score + bias); weights
// from unbiased scores, normalized, rounded through bf16. Dispatch slots via
// atomics (order-invariant: combine is a gather, values don't depend on slot).
// ---------------------------------------------------------------------------
__global__ __launch_bounds__(32) void moe_router_kernel(
    const bf16* __restrict__ X, const bf16* __restrict__ GW,
    const float* __restrict__ bias,
    int* __restrict__ cnt, int* __restrict__ slot2row,
    int* __restrict__ tk_slot, float* __restrict__ tk_w) {
  const int t = blockIdx.x;
  const int e = threadIdx.x;  // lane == expert
  const bf16* xr = X + (size_t)t * DDIM;
  const bf16* gr = GW + (size_t)e * DDIM;
  float acc = 0.f;
#pragma unroll 4
  for (int d = 0; d < DDIM; d += 8) {
    v8bf xv = *(const v8bf*)(xr + d);
    v8bf gv = *(const v8bf*)(gr + d);
#pragma unroll
    for (int j = 0; j < 8; ++j) acc += (float)xv[j] * (float)gv[j];
  }
  const float s = 1.f / (1.f + __expf(-acc));
  float selv = s + bias[e];

  int sel_e[KSEL];
  float sel_w[KSEL];
#pragma unroll
  for (int k = 0; k < KSEL; ++k) {
    float v = selv;
    int ei = e;
#pragma unroll
    for (int off = 16; off > 0; off >>= 1) {
      float ov = __shfl_xor(v, off, 32);
      int oe = __shfl_xor(ei, off, 32);
      if (ov > v || (ov == v && oe < ei)) { v = ov; ei = oe; }
    }
    sel_e[k] = ei;
    sel_w[k] = __shfl(s, ei, 32);
    if (e == ei) selv = -1e30f;
  }
  float wsum = sel_w[0] + sel_w[1] + sel_w[2] + sel_w[3];
  wsum = fmaxf(wsum, 1e-12f);
  if (e < KSEL) {
    const int ex = sel_e[e];
    const float w = (float)(bf16)(sel_w[e] / wsum);
    const int pos = atomicAdd(&cnt[ex], 1);
    int slot = -1;
    if (pos < CAP) {
      slot = ex * CAP + pos;
      slot2row[slot] = t;
    }
    tk_slot[t * KSEL + e] = slot;
    tk_w[t * KSEL + e] = w;
  }
}

// ---------------------------------------------------------------------------
// Tiled WMMA GEMM, 128x128 block, BK=32, 8 waves (4x2), 32x64 per wave.
// GATHER: A rows come from X via slot2row (rows >= cnt[e] are zero).
// BNK:    B stored [N][K] row-major (shared-expert weights); both A and B
//         tiles are DMA'd into LDS by the Tensor Data Mover (wave 0 issues,
//         s_wait_tensorcnt + barrier publishes to the workgroup).
//         else B stored [K][N] (expert weights) -> VGPR transpose into LDS.
// FUSED:  C = silu(A@B1) * (A@B3), else C = A@B1. Output bf16.
// ---------------------------------------------------------------------------
#define BM 128
#define BN 128
#define BK 32
#define LDP 40  // padded LDS stride (elems); 80B rows keep 16B alignment

template <bool GATHER, bool BNK, bool FUSED>
__global__ __launch_bounds__(256) void ffn_gemm_kernel(
    const bf16* __restrict__ A, int lda, long aStride,
    const int* __restrict__ slot2row, const int* __restrict__ cnt,
    const bf16* __restrict__ B1, const bf16* __restrict__ B3, int ldb,
    long bStride, bf16* __restrict__ Cout, int ldc, long cStride, int Kdim) {
  const int z = blockIdx.z;
  const int mbase = blockIdx.y * BM;
  const int nbase = blockIdx.x * BN;
  if (cnt != nullptr && mbase >= cnt[z]) return;  // skip empty expert tiles

  __shared__ bf16 sm[BM * LDP * (FUSED ? 3 : 2)];
  bf16* As = sm;
  bf16* B1t = sm + BM * LDP;
  bf16* B3t = sm + 2 * BM * LDP;  // only touched when FUSED

  const int tid = threadIdx.x;
  const int lane = tid & 31;
  const int wave = tid >> 5;
  const int wm = wave >> 1;   // 0..3
  const int wn = wave & 1;    // 0..1
  const int lc = lane & 15;
  const int lh = lane >> 4;

  // global->LDS thread mapping (row-pair, 16B column chunk)
  const int ar = tid >> 2;         // 0..63
  const int ac = (tid & 3) * 8;    // 0,8,16,24
  const int bkr = tid >> 4;        // 0..15 (KN path)
  const int bnc = (tid & 15) * 8;  // KN path col chunk

  const bf16* aRow[2] = {nullptr, nullptr};
  if (!BNK) {
#pragma unroll
    for (int i = 0; i < 2; ++i) {
      const int r = ar + i * 64;
      if (GATHER) {
        const bool ok = (mbase + r) < cnt[z];
        aRow[i] = ok ? (A + (size_t)slot2row[(size_t)z * CAP + mbase + r] * lda)
                     : nullptr;
      } else {
        aRow[i] = A + (size_t)z * aStride + (size_t)(mbase + r) * lda;
      }
    }
  }

  // LDS byte offsets for TDM (low 32 bits of flat shared pointer)
  const unsigned asOff = (unsigned)(size_t)(void*)As;
  const unsigned b1Off = (unsigned)(size_t)(void*)B1t;
  const unsigned b3Off = (unsigned)(size_t)(void*)B3t;

  v8f accU[2][4] = {};
  v8f accV[2][4] = {};

  for (int k0 = 0; k0 < Kdim; k0 += BK) {
    __syncthreads();
    if (BNK) {
      // ---- TDM path: DMA A + B tile(s) straight into padded LDS
      if (wave == 0) {
        tdm_load_tile_2d(asOff, A + (size_t)mbase * lda + k0, lda);
        tdm_load_tile_2d(b1Off, B1 + (size_t)nbase * ldb + k0, ldb);
        if (FUSED) tdm_load_tile_2d(b3Off, B3 + (size_t)nbase * ldb + k0, ldb);
        __builtin_amdgcn_s_wait_tensorcnt(0);
      }
    } else {
      // ---- A tile -> LDS [m][k]
#pragma unroll
      for (int i = 0; i < 2; ++i) {
        const int r = ar + i * 64;
        v8bf av = {};
        if (aRow[i]) av = *(const v8bf*)(aRow[i] + k0 + ac);
        *(v8bf*)(As + r * LDP + ac) = av;
      }
      // prefetch next k-chunk of gathered A rows (no inter-iter locality)
      if (k0 + BK < Kdim) {
        if (aRow[0]) __builtin_prefetch(aRow[0] + k0 + BK + ac);
        if (aRow[1]) __builtin_prefetch(aRow[1] + k0 + BK + ac);
      }
      // ---- B tile(s) [K][N] -> transpose into LDS Bt[n][k]
#pragma unroll
      for (int i = 0; i < 2; ++i) {
        const int kr = bkr + i * 16;
        const size_t off =
            (size_t)z * bStride + (size_t)(k0 + kr) * ldb + nbase + bnc;
        v8bf b1v = *(const v8bf*)(B1 + off);
#pragma unroll
        for (int j = 0; j < 8; ++j) B1t[(bnc + j) * LDP + kr] = b1v[j];
        if (FUSED) {
          v8bf b3v = *(const v8bf*)(B3 + off);
#pragma unroll
          for (int j = 0; j < 8; ++j) B3t[(bnc + j) * LDP + kr] = b3v[j];
        }
      }
    }
    __syncthreads();

    // ---- fragments + WMMA
    v16bf af[2];
#pragma unroll
    for (int m = 0; m < 2; ++m) {
      const bf16* p = As + (wm * 32 + m * 16 + lc) * LDP;
      af[m] = cat8(*(const v8bf*)(p + 16 * lh), *(const v8bf*)(p + 16 * lh + 8));
    }
#pragma unroll
    for (int n = 0; n < 4; ++n) {
      const bf16* p1 = B1t + (wn * 64 + n * 16 + lc) * LDP + 16 * lh;
      v16bf b1f = cat8(*(const v8bf*)p1, *(const v8bf*)(p1 + 8));
#pragma unroll
      for (int m = 0; m < 2; ++m) accU[m][n] = wmma_bf16(af[m], b1f, accU[m][n]);
      if (FUSED) {
        const bf16* p3 = B3t + (wn * 64 + n * 16 + lc) * LDP + 16 * lh;
        v16bf b3f = cat8(*(const v8bf*)p3, *(const v8bf*)(p3 + 8));
#pragma unroll
        for (int m = 0; m < 2; ++m)
          accV[m][n] = wmma_bf16(af[m], b3f, accV[m][n]);
      }
    }
  }

  // ---- epilogue (C layout: VGPR r -> row r + 8*(lane/16), col = lane%16)
#pragma unroll
  for (int m = 0; m < 2; ++m)
#pragma unroll
    for (int n = 0; n < 4; ++n)
#pragma unroll
      for (int r = 0; r < 8; ++r) {
        const int row = mbase + wm * 32 + m * 16 + lh * 8 + r;
        const int col = nbase + wn * 64 + n * 16 + lc;
        float val = accU[m][n][r];
        if (FUSED) {
          const float u = val;
          val = (u / (1.f + __expf(-u))) * accV[m][n][r];
        }
        Cout[(size_t)z * cStride + (size_t)row * ldc + col] = (bf16)val;
      }
}

// ---------------------------------------------------------------------------
// Combine: out[t] = Ys[t] + sum_k w[t,k] * Ye[slot(t,k)]
// ---------------------------------------------------------------------------
__global__ __launch_bounds__(256) void moe_combine_kernel(
    const bf16* __restrict__ Ys, const bf16* __restrict__ Ye,
    const int* __restrict__ tk_slot, const float* __restrict__ tk_w,
    bf16* __restrict__ out) {
  const int t = blockIdx.x;
  int slot[KSEL];
  float w[KSEL];
#pragma unroll
  for (int k = 0; k < KSEL; ++k) {
    slot[k] = tk_slot[t * KSEL + k];
    w[k] = tk_w[t * KSEL + k];
  }
  const int d0 = threadIdx.x * 4;
#pragma unroll
  for (int j = 0; j < 4; ++j) {
    const int d = d0 + j;
    float acc = (float)Ys[(size_t)t * DDIM + d];
#pragma unroll
    for (int k = 0; k < KSEL; ++k)
      if (slot[k] >= 0) acc += w[k] * (float)Ye[(size_t)slot[k] * DDIM + d];
    out[(size_t)t * DDIM + d] = (bf16)acc;
  }
}

// ---------------------------------------------------------------------------
extern "C" void kernel_launch(void* const* d_in, const int* in_sizes, int n_in,
                              void* d_out, int out_size, void* d_ws,
                              size_t ws_size, hipStream_t stream) {
  (void)in_sizes; (void)n_in; (void)out_size; (void)ws_size;
  const bf16* X = (const bf16*)d_in[0];
  const bf16* GW = (const bf16*)d_in[1];
  const float* bias = (const float*)d_in[2];
  const bf16* W1 = (const bf16*)d_in[3];
  const bf16* W3 = (const bf16*)d_in[4];
  const bf16* W2 = (const bf16*)d_in[5];
  const bf16* SW1 = (const bf16*)d_in[6];
  const bf16* SW3 = (const bf16*)d_in[7];
  const bf16* SW2 = (const bf16*)d_in[8];
  bf16* out = (bf16*)d_out;

  char* ws = (char*)d_ws;
  size_t off = 0;
  auto alloc = [&](size_t bytes) -> void* {
    off = (off + 255) & ~(size_t)255;
    void* p = ws + off;
    off += bytes;
    return p;
  };
  int* tk_slot = (int*)alloc((size_t)NTOK * KSEL * sizeof(int));
  float* tk_w = (float*)alloc((size_t)NTOK * KSEL * sizeof(float));
  int* cnt = (int*)alloc((size_t)NEXP * sizeof(int));
  int* s2r = (int*)alloc((size_t)NEXP * CAP * sizeof(int));
  bf16* H = (bf16*)alloc((size_t)NEXP * CAP * IDIM * sizeof(bf16));
  bf16* Ye = (bf16*)alloc((size_t)NEXP * CAP * DDIM * sizeof(bf16));
  bf16* Hs = (bf16*)alloc((size_t)NTOK * IDIM * sizeof(bf16));
  bf16* Ys = (bf16*)alloc((size_t)NTOK * DDIM * sizeof(bf16));

  (void)hipMemsetAsync(cnt, 0, NEXP * sizeof(int), stream);

  moe_router_kernel<<<NTOK, 32, 0, stream>>>(X, GW, bias, cnt, s2r, tk_slot,
                                             tk_w);

  // Expert FFN1: H = silu(Xe@W1) * (Xe@W3); A gathered, B in [K][N] layout.
  ffn_gemm_kernel<true, false, true>
      <<<dim3(IDIM / BN, CAP / BM, NEXP), 256, 0, stream>>>(
          X, DDIM, 0, s2r, cnt, W1, W3, IDIM, (long)DDIM * IDIM, H, IDIM,
          (long)CAP * IDIM, DDIM);

  // Shared FFN1: Hs = silu(X@sw1^T) * (X@sw3^T); TDM-fed, B in [N][K].
  ffn_gemm_kernel<false, true, true>
      <<<dim3(IDIM / BN, NTOK / BM, 1), 256, 0, stream>>>(
          X, DDIM, 0, nullptr, nullptr, SW1, SW3, DDIM, 0, Hs, IDIM, 0, DDIM);

  // Expert FFN2: Ye = H @ W2; B in [K][N] layout.
  ffn_gemm_kernel<false, false, false>
      <<<dim3(DDIM / BN, CAP / BM, NEXP), 256, 0, stream>>>(
          H, IDIM, (long)CAP * IDIM, nullptr, cnt, W2, nullptr, DDIM,
          (long)IDIM * DDIM, Ye, DDIM, (long)CAP * DDIM, IDIM);

  // Shared FFN2: Ys = Hs @ sw2^T; TDM-fed, B in [N][K].
  ffn_gemm_kernel<false, true, false>
      <<<dim3(DDIM / BN, NTOK / BM, 1), 256, 0, stream>>>(
          Hs, IDIM, 0, nullptr, nullptr, SW2, nullptr, IDIM, 0, Ys, DDIM, 0,
          IDIM);

  moe_combine_kernel<<<NTOK, 256, 0, stream>>>(Ys, Ye, tk_slot, tk_w, out);
}
